// GroundPropagation_74165495267534
// MI455X (gfx1250) — compile-verified, matrix operand order checked by reading, not run on hardware
//
#include <hip/hip_runtime.h>
#include <hip/hip_bf16.h>
#include <math.h>

typedef __attribute__((ext_vector_type(16))) _Float16 v16h;
typedef __attribute__((ext_vector_type(8)))  float    v8f;

#define BB 8
#define CC 128
#define HH 96
#define WW 320
#define HW (HH*WW)
#define CHW (CC*HW)
#define NSEL 32
#define GROUPS 8          // 128 channels / 16
#define EPSV 1e-6f
#define CLIPR 0.3f

__device__ __forceinline__ float disp_of_h(int h) {
    return 0.1f + 0.9f * (float)h / 95.0f;   // linspace(0.1, 1.0, 96)
}

// fast sigmoid: v_exp_f32 + v_rcp_f32 (result truncated to f16 anyway)
__device__ __forceinline__ float fast_sigmoid(float x) {
    return __builtin_amdgcn_rcpf(1.0f + __expf(-x));
}

// ---------------------------------------------------------------- init
__global__ void k_init(float* g_dot, float* g_sum, float* g_ssq,
                       int* idx, unsigned int* Mmax) {
    int t = threadIdx.x;
    if (t < 128) { g_dot[t] = 0.f; g_sum[t] = 0.f; g_ssq[t] = 0.f; }
    if (t < 32)  idx[t] = 0;
    if (t < 256) Mmax[t] = 0u;
}

// --------------------------------------------- fused copy + WMMA reductions
// One wave handles one (b, channel-group, row). Per 32-pixel chunk:
//   acc1 += A(16ch x 32px, f16 sigmoid) x B(32x16)  col0=disp[h], col1=1
//   acc2 += A2(s^2)                     x B2        col0=1
__global__ void k_reduce_wmma(const float* __restrict__ in, float* __restrict__ out,
                              float* g_dot, float* g_sum, float* g_ssq) {
    int wave = threadIdx.x >> 5;
    int lane = threadIdx.x & 31;
    int band = blockIdx.x % 12;              // 12 bands * 8 waves = 96 rows
    int bg   = blockIdx.x / 12;
    int g    = bg & 7;
    int b    = bg >> 3;
    int h    = band * 8 + wave;

    int half = lane >> 4;                    // K-half selector
    int n    = lane & 15;                    // A row (=channel) / B column
    int ch   = g * 16 + n;

    const float* src = in  + (size_t)b * CHW + (size_t)ch * HW + (size_t)h * WW;
    float*       dst = out + (size_t)b * CHW + (size_t)ch * HW + (size_t)h * WW;

    float dh = disp_of_h(h);
    _Float16 bval  = (n == 0) ? (_Float16)dh : ((n == 1) ? (_Float16)1.0f : (_Float16)0.0f);
    _Float16 b2val = (n == 0) ? (_Float16)1.0f : (_Float16)0.0f;
    v16h Bm, B2;
    #pragma unroll
    for (int i = 0; i < 16; ++i) { Bm[i] = bval; B2[i] = b2val; }

    v8f acc1 = {}; v8f acc2 = {};

    for (int chunk = 0; chunk < 10; ++chunk) {       // 10 * 32 = 320 = W
        int p0 = chunk * 32;
        // lane loads K = {half*8+0..7} and {16+half*8+0..7}  (ISA 16-bit A 16x32 layout)
        const float4* s1 = (const float4*)(src + p0 + 8 * half);
        const float4* s2 = (const float4*)(src + p0 + 16 + 8 * half);
        float4 a0 = s1[0], a1 = s1[1], a2 = s2[0], a3 = s2[1];
        // write-through identity copy of the input
        float4* d1 = (float4*)(dst + p0 + 8 * half);
        float4* d2 = (float4*)(dst + p0 + 16 + 8 * half);
        d1[0] = a0; d1[1] = a1; d2[0] = a2; d2[1] = a3;

        float v[16];
        v[0]=a0.x; v[1]=a0.y; v[2]=a0.z; v[3]=a0.w;
        v[4]=a1.x; v[5]=a1.y; v[6]=a1.z; v[7]=a1.w;
        v[8]=a2.x; v[9]=a2.y; v[10]=a2.z; v[11]=a2.w;
        v[12]=a3.x; v[13]=a3.y; v[14]=a3.z; v[15]=a3.w;

        v16h A, A2;
        #pragma unroll
        for (int i = 0; i < 16; ++i) {
            float s = fast_sigmoid(v[i]);
            A[i]  = (_Float16)s;
            A2[i] = (_Float16)(s * s);
        }
        acc1 = __builtin_amdgcn_wmma_f32_16x16x32_f16(false, A,  false, Bm, (short)0, acc1, false, false);
        acc2 = __builtin_amdgcn_wmma_f32_16x16x32_f16(false, A2, false, B2, (short)0, acc2, false, false);
    }

    // D layout: lane (n + 16*hi) VGPR r holds D[r + 8*hi][n]
    // col 0 (lanes 0,16): sum s*disp; col 1 (lanes 1,17): sum s; acc2 col 0: sum s^2
    if (n < 2) {
        float* tgt = (n == 0) ? g_dot : g_sum;
        int base = g * 16 + half * 8;
        #pragma unroll
        for (int r = 0; r < 8; ++r) atomicAdd(&tgt[base + r], acc1[r]);
        if (n == 0) {
            #pragma unroll
            for (int r = 0; r < 8; ++r) atomicAdd(&g_ssq[base + r], acc2[r]);
        }
    }
}

// ---------------------------------------------------- stable top-16 selection
__global__ void k_select(const float* g_dot, const float* g_sum, const float* g_ssq,
                         int* idx_out) {
    __shared__ float cd[128], cz[128];
    int c = threadIdx.x;
    float nd = 0.f, nz = 0.f;
    for (int h = 0; h < HH; ++h) {
        float d = disp_of_h(h);
        nd += d * d;
        nz += (1.f - d) * (1.f - d);
    }
    nd = fmaxf(sqrtf(nd * (float)(BB * WW)), EPSV);
    nz = fmaxf(sqrtf(nz * (float)(BB * WW)), EPSV);
    float sn = fmaxf(sqrtf(g_ssq[c]), EPSV);
    cd[c] = g_dot[c] / (sn * nd);
    cz[c] = (g_sum[c] - g_dot[c]) / (sn * nz);
    __syncthreads();
    float vd = cd[c], vz = cz[c];
    int rd = 0, rz = 0;
    for (int k = 0; k < 128; ++k) {
        rd += (cd[k] > vd) || (cd[k] == vd && k < c);   // stable descending rank
        rz += (cz[k] > vz) || (cz[k] == vz && k < c);
    }
    if (rd < 16) idx_out[rd] = c;
    if (rz < 16) idx_out[16 + rz] = c;
}

// ------------------------------------- collapse 32-step scan into gather index
// j(h) = min(h + min(run_of_ones_from(h), 32), H-1)
__global__ void k_scan(const int* __restrict__ mask, unsigned char* __restrict__ jidx) {
    int b = blockIdx.x, w = threadIdx.x;
    int run = 0;
    for (int h = HH - 1; h >= 0; --h) {
        int m = mask[((size_t)b * HH + h) * WW + w];
        run = m ? (run + 1) : 0;
        int r = run > 32 ? 32 : run;
        int j = h + r; if (j > HH - 1) j = HH - 1;
        jidx[((size_t)b * HH + h) * WW + w] = (unsigned char)j;
    }
}

// ------------------------------------------ per-(b,c) max of |prop - sel|
__global__ void k_maxdiff(const float* __restrict__ in, const int* __restrict__ idx,
                          const unsigned char* __restrict__ jidx, unsigned int* Mmax) {
    __shared__ float red[320];
    int w = threadIdx.x;
    int h = blockIdx.x % HH;
    int t = blockIdx.x / HH;
    int c = t % NSEL, b = t / NSEL;
    int ch = idx[c];
    const float* base = in + (size_t)b * CHW + (size_t)ch * HW;
    float sel  = base[(size_t)h * WW + w];
    int   j    = jidx[((size_t)b * HH + h) * WW + w];
    float prop = base[(size_t)j * WW + w];
    red[w] = fabsf(prop - sel);
    __syncthreads();
    if (w < 64) red[w] = fmaxf(red[w], red[w + 256]);
    __syncthreads();
    for (int s = 128; s > 0; s >>= 1) {
        if (w < s) red[w] = fmaxf(red[w], red[w + s]);
        __syncthreads();
    }
    if (w == 0) atomicMax(&Mmax[b * NSEL + c], __float_as_uint(red[0]));  // diff >= 0
}

// ------------------------------------------------- final blend + scatter
__global__ void k_blend(const float* __restrict__ in, float* __restrict__ out,
                        const int* __restrict__ idx, const unsigned char* __restrict__ jidx,
                        const unsigned int* __restrict__ Mmax) {
    int w = threadIdx.x;
    int h = blockIdx.x % HH;
    int b = blockIdx.x / HH;
    int j = jidx[((size_t)b * HH + h) * WW + w];

    float selv[NSEL], propv[NSEL];
    float wmax = 0.f;
    #pragma unroll 4
    for (int c = 0; c < NSEL; ++c) {
        int ch = idx[c];
        const float* base = in + (size_t)b * CHW + (size_t)ch * HW;
        float s = base[(size_t)h * WW + w];
        float p = base[(size_t)j * WW + w];
        selv[c] = s; propv[c] = p;
        float M = __uint_as_float(Mmax[b * NSEL + c]) * CLIPR;
        if (M == 0.f) M = 1.f;
        float wc = fminf(fabsf(p - s) / M, 1.f);
        wmax = fmaxf(wmax, wc);
    }
    #pragma unroll 4
    for (int c = 0; c < NSEL; ++c) {
        int ch = idx[c];
        out[(size_t)b * CHW + (size_t)ch * HW + (size_t)h * WW + w] =
            wmax * propv[c] + (1.f - wmax) * selv[c];
    }
}

extern "C" void kernel_launch(void* const* d_in, const int* in_sizes, int n_in,
                              void* d_out, int out_size, void* d_ws, size_t ws_size,
                              hipStream_t stream) {
    const float* in   = (const float*)d_in[0];   // (8,128,96,320) f32
    const int*   mask = (const int*)d_in[1];     // (8,96,320) i32
    float*       out  = (float*)d_out;           // (8,128,96,320) f32

    float* g_dot = (float*)d_ws;                       // 128
    float* g_sum = g_dot + 128;                        // 128
    float* g_ssq = g_sum + 128;                        // 128
    int*   idx   = (int*)(g_ssq + 128);                // 32
    unsigned int*  Mmax = (unsigned int*)(idx + 32);   // 256 (8*32)
    unsigned char* jidx = (unsigned char*)(Mmax + 256);// 8*96*320 bytes

    hipLaunchKernelGGL(k_init, dim3(1), dim3(512), 0, stream,
                       g_dot, g_sum, g_ssq, idx, Mmax);
    hipLaunchKernelGGL(k_reduce_wmma, dim3(BB * GROUPS * 12), dim3(256), 0, stream,
                       in, out, g_dot, g_sum, g_ssq);
    hipLaunchKernelGGL(k_select, dim3(1), dim3(128), 0, stream,
                       g_dot, g_sum, g_ssq, idx);
    hipLaunchKernelGGL(k_scan, dim3(BB), dim3(WW), 0, stream, mask, jidx);
    hipLaunchKernelGGL(k_maxdiff, dim3(BB * NSEL * HH), dim3(WW), 0, stream,
                       in, idx, jidx, Mmax);
    hipLaunchKernelGGL(k_blend, dim3(BB * HH), dim3(WW), 0, stream,
                       in, out, idx, jidx, Mmax);
}
